// ScannedLSTM_36352603193854
// MI455X (gfx1250) — matrix operand order repaired and checked by reading, not compiled, not run on hardware
//
#include <hip/hip_runtime.h>
#include <stdint.h>

// Scanned LSTM on gfx1250 (MI455X), wave32 + WMMA bf16.
// B=512 batch tiles of 16 rows -> 32 workgroups x 4 waves.
// Wave w owns h-columns [16w,16w+16) == one 16-wide strip of each gate.
// Weights [Wi;Wh] (128x256) preloaded as bf16 B-fragments into VGPRs (128 VGPRs).
// A-tile per step = [x_t | h] (16x128 bf16) staged in LDS; x_{t+1}/resets_{t+1}
// are software-pipelined (prefetched during step t's compute phase).
// Gates use v_exp_f32 + v_rcp_f32 (TRANS pipe) -- no IEEE division sequences.

typedef __attribute__((ext_vector_type(16))) __bf16         v16bf;
typedef __attribute__((ext_vector_type(16))) unsigned short v16us;
typedef __attribute__((ext_vector_type(8)))  float          v8f;
typedef __attribute__((ext_vector_type(4)))  unsigned int   v4u;

#define BSZ 512
#define TSZ 512
#define FSZ 64
#define HSZ 64
#define NG  256            // 4*H
#define XH_STRIDE 136      // halves per A row: 128 + 8 pad (bank-conflict avoidance)

struct V4x2 { v4u lo, hi; };   // 32 bytes == v16bf

__device__ __forceinline__ unsigned short f2bf(float f) {
  unsigned int u = __float_as_uint(f);
  u += 0x7FFFu + ((u >> 16) & 1u);      // round-to-nearest-even
  return (unsigned short)(u >> 16);
}
// Fast gate functions: v_exp_f32 + v_rcp_f32, no div-scale/div-fixup sequences.
__device__ __forceinline__ float sigm(float x) {
  return __builtin_amdgcn_rcpf(1.0f + __expf(-x));
}
__device__ __forceinline__ float tanh_fast(float x) {
  return 2.0f * __builtin_amdgcn_rcpf(1.0f + __expf(-2.0f * x)) - 1.0f;
}

__global__ void __launch_bounds__(128, 1)
lstm_scan_kernel(const float* __restrict__ ins,
                 const unsigned char* __restrict__ resets,
                 const float* __restrict__ Wi,
                 const float* __restrict__ Wh,
                 const float* __restrict__ bias,
                 float* __restrict__ out) {
  __shared__ alignas(16) unsigned short sXH[16 * XH_STRIDE];  // [row][0..63]=x, [64..127]=h (bf16)
  __shared__ alignas(8)  unsigned int   sFlagW[4];            // 16 reset bytes

  const int tid = threadIdx.x;
  const int w   = tid >> 5;        // wave id 0..3
  const int l   = tid & 31;        // lane
  const int lhi = l >> 4;          // 0/1 (half-wave)
  const int llo = l & 15;
  const int b0  = blockIdx.x * 16; // batch-tile base row

  // ---- Preload weight B-fragments into VGPRs: bF[gate j][kchunk kc]
  // B 32x16 bf16 layout: lanes 0-15 hold K=k0..k0+7,k0+16..k0+23 of column N=llo; lanes 16-31 K=k0+8..k0+15,k0+24..k0+31.
  v16bf bF[4][4];
  #pragma unroll
  for (int j = 0; j < 4; ++j) {
    const int N = 16 * w + 64 * j + llo;
    #pragma unroll
    for (int kc = 0; kc < 4; ++kc) {
      const int k0 = kc * 32 + 8 * lhi;
      v16us tmp;
      #pragma unroll
      for (int kk = 0; kk < 16; ++kk) {
        const int k = k0 + kk + (kk >= 8 ? 8 : 0);
        float v = (kc < 2) ? Wi[k * NG + N] : Wh[(k - 64) * NG + N];
        tmp[kk] = f2bf(v);
      }
      bF[j][kc] = __builtin_bit_cast(v16bf, tmp);
    }
  }
  // Bias broadcast value per gate (same for every row of the C tile)
  float bv[4];
  #pragma unroll
  for (int j = 0; j < 4; ++j) bv[j] = bias[16 * w + 64 * j + llo];

  // Zero the h region of the A staging buffer (h0 = 0)
  const int xr = tid >> 3, xs = tid & 7;           // x-staging: row, 8-float chunk
  {
    v4u z = {0u, 0u, 0u, 0u};
    *reinterpret_cast<v4u*>(&sXH[xr * XH_STRIDE + 64 + xs * 8]) = z;
  }

  // Carried state in registers (C/D layout: element i -> row M=i+8*lhi, col N=16w+llo)
  v8f cst;
  float ht[8];
  #pragma unroll
  for (int i = 0; i < 8; ++i) { cst[i] = 0.0f; ht[i] = 0.0f; }

  const int col = 16 * w + llo;    // this lane's h column

  // ---- Software-pipelined prefetch of x_t / resets_t (t = 0)
  const float* xsrc = ins + (size_t)(b0 + xr) * TSZ * FSZ + xs * 8;
  const unsigned char* rsrc = resets + (size_t)(b0 + (tid & 15)) * TSZ;
  float4 px0 = *reinterpret_cast<const float4*>(xsrc);
  float4 px1 = *reinterpret_cast<const float4*>(xsrc + 4);
  unsigned char pfl = (tid < 16) ? rsrc[0] : (unsigned char)0;

  for (int t = 0; t < TSZ; ++t) {
    __syncthreads();  // prior step's A-fragment reads complete; safe to restage

    // ---- Stage phase: flags, x_t (fp32 -> bf16), previous h -- all from registers
    if (tid < 16)
      reinterpret_cast<unsigned char*>(sFlagW)[tid] = pfl;
    {
      v4u pk;
      pk.x = (unsigned)f2bf(px0.x) | ((unsigned)f2bf(px0.y) << 16);
      pk.y = (unsigned)f2bf(px0.z) | ((unsigned)f2bf(px0.w) << 16);
      pk.z = (unsigned)f2bf(px1.x) | ((unsigned)f2bf(px1.y) << 16);
      pk.w = (unsigned)f2bf(px1.z) | ((unsigned)f2bf(px1.w) << 16);
      *reinterpret_cast<v4u*>(&sXH[xr * XH_STRIDE + xs * 8]) = pk;
    }
    #pragma unroll
    for (int i = 0; i < 8; ++i) {       // write h_{t-1} (bf16) into A buffer
      const int M = i + 8 * lhi;
      sXH[M * XH_STRIDE + 64 + col] = f2bf(ht[i]);
    }
    __syncthreads();

    // ---- Prefetch x_{t+1} / resets_{t+1}; latency hidden behind WMMA + gates
    {
      const int tn = (t + 1 < TSZ) ? (t + 1) : t;
      px0 = *reinterpret_cast<const float4*>(xsrc + (size_t)tn * FSZ);
      px1 = *reinterpret_cast<const float4*>(xsrc + (size_t)tn * FSZ + 4);
      if (tid < 16) pfl = rsrc[tn];
    }

    // ---- Reset masks (32-bit flag words)
    const unsigned int fw0 = sFlagW[2 * lhi];
    const unsigned int fw1 = sFlagW[2 * lhi + 1];
    const unsigned int flA = reinterpret_cast<const unsigned char*>(sFlagW)[llo]; // flag for A row M=llo
    const unsigned int hmask = flA ? 0u : 0xFFFFFFFFu;
    #pragma unroll
    for (int i = 0; i < 8; ++i) {
      const unsigned int fb = ((i < 4 ? fw0 : fw1) >> (8 * (i & 3))) & 0xFFu;
      cst[i] = fb ? 0.0f : cst[i];
    }

    // ---- A fragments (16-bit A 16x32 layout), zero h-half on reset
    v16bf aF[4];
    #pragma unroll
    for (int kc = 0; kc < 4; ++kc) {
      const int off = llo * XH_STRIDE + kc * 32 + 8 * lhi;
      V4x2 p;
      p.lo = *reinterpret_cast<const v4u*>(&sXH[off]);
      p.hi = *reinterpret_cast<const v4u*>(&sXH[off + 16]);
      if (kc >= 2) { p.lo &= hmask; p.hi &= hmask; }
      aF[kc] = __builtin_bit_cast(v16bf, p);
    }

    // ---- z = [x|h] @ [Wi;Wh] + b : 4 independent WMMA chains (one per gate)
    v8f z[4];
    #pragma unroll
    for (int j = 0; j < 4; ++j) {
      v8f acc;
      #pragma unroll
      for (int i = 0; i < 8; ++i) acc[i] = bv[j];
      #pragma unroll
      for (int kc = 0; kc < 4; ++kc)
        acc = __builtin_amdgcn_wmma_f32_16x16x32_bf16(
            false, aF[kc], false, bF[j][kc], (short)0, acc, false, false);
      z[j] = acc;
    }

    // ---- Gates, state update, output store
    #pragma unroll
    for (int i = 0; i < 8; ++i) {
      const float ig = sigm(z[0][i]);
      const float fg = sigm(z[1][i]);
      const float gg = tanh_fast(z[2][i]);
      const float og = sigm(z[3][i]);
      const float cn = fg * cst[i] + ig * gg;
      cst[i] = cn;
      ht[i]  = og * tanh_fast(cn);
      const int M = i + 8 * lhi;
      out[((size_t)(b0 + M) * TSZ + t) * HSZ + col] = ht[i];
    }
  }
}

extern "C" void kernel_launch(void* const* d_in, const int* in_sizes, int n_in,
                              void* d_out, int out_size, void* d_ws, size_t ws_size,
                              hipStream_t stream) {
  const float*         ins    = (const float*)d_in[0];
  const unsigned char* resets = (const unsigned char*)d_in[1];  // jnp bool -> 1 byte
  const float*         Wi     = (const float*)d_in[2];
  const float*         Wh     = (const float*)d_in[3];
  const float*         bias   = (const float*)d_in[4];
  float*               out    = (float*)d_out;

  dim3 grid(BSZ / 16);   // 32 workgroups, one 16-row batch tile each
  dim3 block(128);       // 4 waves
  lstm_scan_kernel<<<grid, block, 0, stream>>>(ins, resets, Wi, Wh, bias, out);
}
